// PCNN_23003844838098
// MI455X (gfx1250) — matrix operand order repaired and checked by reading, not compiled, not run on hardware
//
#include <hip/hip_runtime.h>

// PCNN fused kernel for MI455X (gfx1250, wave32, WMMA).
// Strategy: per batch item, conv1d(K=3) as GEMM  out[128][512] = W[128x384] x im2col(x)[384x512]
// with K reordered as (k*128 + e) so im2col rows are contiguous embedding rows.
// Fused: emb gather -> bf16 convert -> WMMA conv -> bias+ReLU -> piecewise max-pool -> FC.

typedef __attribute__((ext_vector_type(16))) __bf16 v16bf;
typedef __attribute__((ext_vector_type(8)))  float  v8f;

#define LSEQ  512
#define EMBD  128
#define FILT  128
#define KW    3
#define NC    53
#define CHUNK 128          // positions per LDS staging chunk
#define KSTEPS 12          // 384 / 32

__global__ __launch_bounds__(256, 1) void pcnn_fused_kernel(
    const int*   __restrict__ char_ids,   // (B, 512) int32
    const int*   __restrict__ pos_e1,     // (B,)
    const int*   __restrict__ pos_e2,     // (B,)
    const float* __restrict__ emb_table,  // (21128, 128)
    const float* __restrict__ conv_w,     // (128, 128, 3)  [F][E][K]
    const float* __restrict__ conv_b,     // (128,)
    const float* __restrict__ fc_w,       // (53, 384)
    const float* __restrict__ fc_b,       // (53,)
    float*       __restrict__ out)        // (B, 53)
{
    // Embedding chunk (+1 halo each side), bf16, row-major [pos][e]; row = 256B.
    __shared__ __bf16 sX[(CHUNK + 2) * EMBD];   // 33280 B
    __shared__ int    pool[3 * FILT];           // segment maxima as float bits (>=0)

    const int b    = blockIdx.x;
    const int tid  = threadIdx.x;
    const int wave = tid >> 5;      // 0..7 -> mtile (16 filters each)
    const int lane = tid & 31;
    const int half = lane >> 4;     // lane half selects K sub-block / D row offset
    const int ln16 = lane & 15;

    // ---- entity segment bounds (uniform per block)
    int p1 = pos_e1[b], p2 = pos_e2[b];
    int e1 = min(p1, p2), e2 = max(p1, p2);
    e1 = min(max(e1, 0), LSEQ);
    e2 = min(max(e2, 0), LSEQ);
    if (e1 == e2) e2 = min(e1 + 1, LSEQ);
    const int m1b = max(e1, 1);          // seg1: l <  m1b
    const int m3b = min(e2, LSEQ - 1);   // seg3: l >= m3b

    for (int i = tid; i < 3 * FILT; i += blockDim.x) pool[i] = 0;  // 0 bits == 0.0f

    // ---- A fragments (weights) register-resident, 16-bit A 16x32 fragments in this kernel's own K order:
// row M = lane&15; element i holds K = half*16 + i (contiguous, matching B).
    const int mtile = wave;
    const int fA    = mtile * 16 + ln16;
    v16bf a[KSTEPS];
    #pragma unroll
    for (int ks = 0; ks < KSTEPS; ++ks) {
        #pragma unroll
        for (int i = 0; i < 16; ++i) {
            int K  = ks * 32 + half * 16 + i;
            int e  = K & (EMBD - 1);   // reordered K = k*128 + e
            int kk = K >> 7;
            a[ks][i] = (__bf16)conv_w[(fA * EMBD + e) * KW + kk];
        }
    }

    // D layout: lanes<16 -> M=r, lanes>=16 -> M=r+8 ; N = lane&15.
    float bias_r[8];
    #pragma unroll
    for (int r = 0; r < 8; ++r) bias_r[r] = conv_b[mtile * 16 + half * 8 + r];

    float regm[8][3];
    #pragma unroll
    for (int r = 0; r < 8; ++r) { regm[r][0] = 0.f; regm[r][1] = 0.f; regm[r][2] = 0.f; }

    for (int chunk = 0; chunk < LSEQ / CHUNK; ++chunk) {
        __syncthreads();
        // ---- stage embeddings: rows rr = 0..129 map to position chunk*128 - 1 + rr
        for (int rr = (tid >> 1); rr < CHUNK + 2; rr += (blockDim.x >> 1)) {
            const int gl      = chunk * CHUNK - 1 + rr;
            const int colbase = (tid & 1) * 64;
            __bf16* dst = &sX[rr * EMBD + colbase];
            if (gl >= 0 && gl < LSEQ) {
                const int id = char_ids[b * LSEQ + gl];
                const float4* src = (const float4*)&emb_table[id * EMBD + colbase];
                #pragma unroll
                for (int q = 0; q < 16; ++q) {
                    float4 v = src[q];
                    dst[q * 4 + 0] = (__bf16)v.x;
                    dst[q * 4 + 1] = (__bf16)v.y;
                    dst[q * 4 + 2] = (__bf16)v.z;
                    dst[q * 4 + 3] = (__bf16)v.w;
                }
            } else {
                #pragma unroll
                for (int q = 0; q < 64; ++q) dst[q] = (__bf16)0.f;
            }
        }
        __syncthreads();

        // ---- 8 N-tiles of 16 positions per chunk; 12 WMMAs per tile (K=384)
        for (int nt = 0; nt < CHUNK / 16; ++nt) {
            v8f acc = {};
            #pragma unroll
            for (int ks = 0; ks < KSTEPS; ++ks) {
                // B 32x16 bf16 layout: col = lane&15; lanes<16 K=0..15, lanes>=16 K=16..31.
                // Reordered K: conv tap k = ks>>2 shifts the position row; e-offset contiguous.
                const int srow = nt * 16 + ln16 + (ks >> 2);            // +halo handled by row 0 = pos-1
                const int off  = (ks & 3) * 32 + half * 16;             // 16 contiguous bf16, 32B aligned
                const v16bf bfrag = *(const v16bf*)&sX[srow * EMBD + off];
                acc = __builtin_amdgcn_wmma_f32_16x16x32_bf16(
                          false, a[ks], false, bfrag, (short)0, acc, false, false);
            }
            const int lpos = chunk * CHUNK + nt * 16 + ln16;
            #pragma unroll
            for (int r = 0; r < 8; ++r) {
                float v = fmaxf(acc[r] + bias_r[r], 0.f);
                if (lpos < m1b)                regm[r][0] = fmaxf(regm[r][0], v);
                if (lpos >= e1 && lpos < e2)   regm[r][1] = fmaxf(regm[r][1], v);
                if (lpos >= m3b)               regm[r][2] = fmaxf(regm[r][2], v);
            }
        }
    }

    // ---- combine per-lane maxima (all values >= 0, so float bits order as ints)
    {
        const int fbase = mtile * 16 + half * 8;
        #pragma unroll
        for (int r = 0; r < 8; ++r)
            #pragma unroll
            for (int s = 0; s < 3; ++s)
                atomicMax(&pool[s * FILT + fbase + r], __float_as_int(regm[r][s]));
    }
    __syncthreads();

    // ---- tiny FC: out[b][c] = pooled . fc_w[c] + fc_b[c]
    for (int c = tid; c < NC; c += blockDim.x) {
        float accf = fc_b[c];
        const float* wrow = &fc_w[c * (3 * FILT)];
        #pragma unroll 4
        for (int j = 0; j < 3 * FILT; ++j)
            accf += __int_as_float(pool[j]) * wrow[j];
        out[b * NC + c] = accf;
    }
}

extern "C" void kernel_launch(void* const* d_in, const int* in_sizes, int n_in,
                              void* d_out, int out_size, void* d_ws, size_t ws_size,
                              hipStream_t stream) {
    const int*   char_ids = (const int*)d_in[0];
    const int*   pos_e1   = (const int*)d_in[1];
    const int*   pos_e2   = (const int*)d_in[2];
    const float* emb      = (const float*)d_in[3];
    const float* conv_w   = (const float*)d_in[4];
    const float* conv_b   = (const float*)d_in[5];
    const float* fc_w     = (const float*)d_in[6];
    const float* fc_b     = (const float*)d_in[7];
    float* out = (float*)d_out;

    const int B = in_sizes[1];   // pos_e1 has one entry per batch item
    pcnn_fused_kernel<<<B, 256, 0, stream>>>(char_ids, pos_e1, pos_e2, emb,
                                             conv_w, conv_b, fc_w, fc_b, out);
}
